// GCNModelVAE_36043365548238
// MI455X (gfx1250) — matrix operand order repaired and checked by reading, not compiled, not run on hardware
//
#include <hip/hip_runtime.h>

// ---------------------------------------------------------------------------
// GCN-VAE forward for MI455X (gfx1250): bf16 WMMA GEMMs with double-buffered
// async global->LDS staging (ASYNCcnt), fused masked softmax / adjacency
// accumulation. Wave32, WGP, v_wmma_f32_16x16x32_bf16.
// ---------------------------------------------------------------------------

typedef __attribute__((ext_vector_type(16))) __bf16 v16bf;
typedef __attribute__((ext_vector_type(8)))  float  v8f;

#define TM  64    // block tile M
#define TN  128   // block tile N
#define TKK 32    // K step (one bf16 WMMA K)

union FragBF { v16bf v; __bf16 e[16]; };

// Per-lane async copy of 16 bytes global -> LDS (ASYNCcnt tracked).
// lds_addr: LDS byte address (low 32 bits of generic shared pointer).
__device__ __forceinline__ void async_g2l_b128(uint32_t lds_addr, const void* gaddr)
{
    asm volatile("global_load_async_to_lds_b128 %0, %1, off"
                 :: "v"(lds_addr), "v"(gaddr) : "memory");
}
__device__ __forceinline__ void wait_async0()
{
    asm volatile("s_wait_asynccnt 0x0" ::: "memory");
}

// ---------------------------------------------------------------------------
// C(f32, MxN row-major) = A(bf16, MxK row-major) @ op(B)
//   transB == 0 : B is KxN row-major (manual LDS transpose, bounds-checked)
//   transB == 1 : B is NxK row-major (C = A @ B^T), fully async-staged
// 8 waves/block; each wave computes 32x32 via 2x2 16x16x32 bf16 WMMA tiles.
// Double-buffered LDS: async copy of tile t+1 overlaps WMMA on tile t, with a
// SINGLE workgroup barrier per K-step (a wave's fragment ds_loads complete
// before its WMMAs issue, which precede its next barrier arrival; writers of a
// buffer only run after that barrier).
// NOTE: M is always a multiple of TM and (transB==1) N a multiple of TN in
// every call below, so the async paths never need zero-fill.
// ---------------------------------------------------------------------------
__global__ __launch_bounds__(256)
void k_gemm_wmma(const __bf16* __restrict__ A, const __bf16* __restrict__ B,
                 float* __restrict__ C, int M, int N, int K,
                 int transB, int relu)
{
    __shared__ __align__(16) __bf16 Asl[2][TM * TKK];   // 2 x 4 KB
    __shared__ __align__(16) __bf16 Bsl[2][TN * TKK];   // 2 x 8 KB, stored [n][k]

    const int tid   = threadIdx.x;
    const int lane  = tid & 31;
    const int wid   = tid >> 5;
    const int wm    = (wid >> 2) * 32;   // wave row base within block tile
    const int wn    = (wid & 3)  * 32;   // wave col base within block tile
    const int tileM = blockIdx.y * TM;
    const int tileN = blockIdx.x * TN;

    // staging coordinates (fixed per thread)
    const int ar = tid >> 2;             // A row within tile  (0..63)
    const int ac = (tid & 3) << 3;       // A col within tile  (0,8,16,24)
    const int br = tid >> 1;             // B^T row (n) within tile (0..127)
    const int bc = (tid & 1) << 4;       // B^T col (k) within tile (0,16)
    const int bk = tid >> 3;             // B k within tile (0..31)
    const int bn0 = (tid & 7) << 4;      // B n base within tile (0..112)

    const v8f vzero = {0.f,0.f,0.f,0.f,0.f,0.f,0.f,0.f};
    v8f acc[2][2];
    acc[0][0] = vzero; acc[0][1] = vzero; acc[1][0] = vzero; acc[1][1] = vzero;

    // A fragment lane map (ISA 7.12.2, 16-bit A 16x32):
    //   lanes 0-15: row M=lane,    K = {0..7, 16..23}
    //   lanes 16-31: row M=lane-16, K = {8..15, 24..31}
    const int lm   = lane & 15;
    const int asel = (lane >> 4) * 8;    // A K-base for this half-wave
    const int bsel = (lane >> 4) * 16;   // unused; B uses same K order as A (lanes 0-15: K {0..7,16..23}; 16-31: K {8..15,24..31})

    auto stage = [&](int k0, int buf) {
        // ---- A tile 64x32: one async b128 per lane ----
        async_g2l_b128((uint32_t)(uintptr_t)&Asl[buf][ar * TKK + ac],
                       A + (size_t)(tileM + ar) * K + k0 + ac);
        if (transB) {
            // B is NxK row-major: two async b128 per lane, conflict-free layout
            const __bf16* g = B + (size_t)(tileN + br) * K + k0 + bc;
            const uint32_t l = (uint32_t)(uintptr_t)&Bsl[buf][br * TKK + bc];
            async_g2l_b128(l,      g);
            async_g2l_b128(l + 16, g + 8);
        } else {
            // B is KxN row-major: coalesced 32B load per thread, transpose into LDS
            const __bf16* p = B + (size_t)(k0 + bk) * N + tileN + bn0;
            #pragma unroll
            for (int j = 0; j < 16; ++j) {
                const int n = bn0 + j;
                __bf16 v = (tileN + n < N) ? p[j] : (__bf16)0.f;
                Bsl[buf][n * TKK + bk] = v;
            }
            if (k0 + TKK < K)   // speculative prefetch of next K tile
                __builtin_prefetch(p + (size_t)TKK * N, 0, 1);
        }
    };

    const int steps = K / TKK;           // K is always a multiple of 32 here
    stage(0, 0);

    for (int t = 0; t < steps; ++t) {
        const int buf = t & 1;
        wait_async0();                   // my async writes (into buf) are done
        __syncthreads();                 // everyone's writes visible; everyone
                                         // has consumed buf^1 fragments
        if (t + 1 < steps)
            stage((t + 1) * TKK, buf ^ 1);   // overlap copy of next tile

        // ---- fragments + 4 WMMAs on current buffer ----
        FragBF af[2], bf[2];
        #pragma unroll
        for (int u = 0; u < 2; ++u) {
            const __bf16* ap = &Asl[buf][(wm + u * 16 + lm) * TKK];
            #pragma unroll
            for (int v = 0; v < 8; ++v) {
                const int k = (v < 4) ? (asel + 2 * v) : (16 + asel + 2 * (v - 4));
                af[u].e[2 * v]     = ap[k];
                af[u].e[2 * v + 1] = ap[k + 1];
            }
            const __bf16* bp = &Bsl[buf][(wn + u * 16 + lm) * TKK];
            #pragma unroll
            for (int v = 0; v < 8; ++v) {
                const int k = (v < 4) ? (asel + 2 * v) : (16 + asel + 2 * (v - 4));
                bf[u].e[2 * v]     = bp[k];
                bf[u].e[2 * v + 1] = bp[k + 1];
            }
        }
        #pragma unroll
        for (int mi = 0; mi < 2; ++mi)
            #pragma unroll
            for (int ni = 0; ni < 2; ++ni)
                acc[mi][ni] = __builtin_amdgcn_wmma_f32_16x16x32_bf16(
                    false, af[mi].v, false, bf[ni].v,
                    (short)0, acc[mi][ni], false, false);
        // no trailing barrier: fragment loads are register-resident before the
        // next iteration's top barrier, and buf is only refilled after it.
    }

    // ---- store f32 C (ISA 7.12.2 C layout: lane n = lane&15, VGPR r -> M=r+8*(lane>>4)) ----
    #pragma unroll
    for (int mi = 0; mi < 2; ++mi) {
        #pragma unroll
        for (int ni = 0; ni < 2; ++ni) {
            const int cn  = tileN + wn + ni * 16 + (lane & 15);
            const int cm0 = tileM + wm + mi * 16 + ((lane >> 4) << 3);
            if (cn < N) {
                #pragma unroll
                for (int r = 0; r < 8; ++r) {
                    const int cm = cm0 + r;
                    if (cm < M) {
                        float v = acc[mi][ni][r];
                        if (relu) v = fmaxf(v, 0.f);
                        C[(size_t)cm * N + cn] = v;
                    }
                }
            }
        }
    }
}

// ---------------------------------------------------------------------------
// Fused masked softmax + adjacency accumulation. One 256-thread block per row.
// adj_i = softmax(where(adj0>0, scores, -9e15));  adj_new += adj_i;
// adj_sum += adj_i (f32 and bf16 shadow).
// ---------------------------------------------------------------------------
__global__ __launch_bounds__(256)
void k_softmax_accum(const float* __restrict__ scores,
                     const float* __restrict__ adj0,
                     float* __restrict__ adj_new,
                     float* __restrict__ adj_sum,
                     __bf16* __restrict__ adj_bf, int N)
{
    __shared__ float srow[4096];
    __shared__ float redm[8];
    __shared__ float reds[8];
    const int row  = blockIdx.x;
    const int tid  = threadIdx.x;
    const int lane = tid & 31;
    const int wid  = tid >> 5;
    const size_t base = (size_t)row * N;

    float m = -3.0e38f;
    for (int c = tid; c < N; c += 256) {
        float s = scores[base + c];
        s = (adj0[base + c] > 0.f) ? s : -9.0e15f;
        srow[c] = s;
        m = fmaxf(m, s);
    }
    #pragma unroll
    for (int off = 16; off > 0; off >>= 1) m = fmaxf(m, __shfl_xor(m, off, 32));
    if (lane == 0) redm[wid] = m;
    __syncthreads();
    if (tid == 0) {
        float mm = redm[0];
        for (int i = 1; i < 8; ++i) mm = fmaxf(mm, redm[i]);
        redm[0] = mm;
    }
    __syncthreads();
    const float rowmax = redm[0];

    float sum = 0.f;
    for (int c = tid; c < N; c += 256) {
        const float e = __expf(srow[c] - rowmax);
        srow[c] = e;
        sum += e;
    }
    #pragma unroll
    for (int off = 16; off > 0; off >>= 1) sum += __shfl_xor(sum, off, 32);
    if (lane == 0) reds[wid] = sum;
    __syncthreads();
    if (tid == 0) {
        float ss = 0.f;
        for (int i = 0; i < 8; ++i) ss += reds[i];
        reds[0] = ss;
    }
    __syncthreads();
    const float inv = 1.0f / reds[0];

    for (int c = tid; c < N; c += 256) {
        const float p = srow[c] * inv;
        adj_new[base + c] += p;
        const float ns = adj_sum[base + c] + p;
        adj_sum[base + c] = ns;
        adj_bf[base + c]  = (__bf16)ns;
    }
}

// ---------------------------------------------------------------------------
// Row-wise log_softmax for the classifier (C small, one thread per row).
// ---------------------------------------------------------------------------
__global__ void k_logsoftmax_row(const float* __restrict__ in,
                                 float* __restrict__ out, int N, int C)
{
    const int row = blockIdx.x * blockDim.x + threadIdx.x;
    if (row >= N) return;
    const float* p = in + (size_t)row * C;
    float m = p[0];
    for (int c = 1; c < C; ++c) m = fmaxf(m, p[c]);
    float s = 0.f;
    for (int c = 0; c < C; ++c) s += __expf(p[c] - m);
    const float ls = __logf(s);
    float* q = out + (size_t)row * C;
    for (int c = 0; c < C; ++c) q[c] = p[c] - m - ls;
}

// ---------------------------------------------------------------------------
// Elementwise helpers (grid-stride).
// ---------------------------------------------------------------------------
__global__ void k_zero_f(float* __restrict__ p, int n) {
    for (int i = blockIdx.x * blockDim.x + threadIdx.x; i < n; i += gridDim.x * blockDim.x)
        p[i] = 0.f;
}
__global__ void k_zero_bf(__bf16* __restrict__ p, int n) {
    for (int i = blockIdx.x * blockDim.x + threadIdx.x; i < n; i += gridDim.x * blockDim.x)
        p[i] = (__bf16)0.f;
}
__global__ void k_f2bf(const float* __restrict__ a, __bf16* __restrict__ b, int n) {
    for (int i = blockIdx.x * blockDim.x + threadIdx.x; i < n; i += gridDim.x * blockDim.x)
        b[i] = (__bf16)a[i];
}
__global__ void k_copy_f2bf(const float* __restrict__ a, float* __restrict__ c,
                            __bf16* __restrict__ b, int n) {
    for (int i = blockIdx.x * blockDim.x + threadIdx.x; i < n; i += gridDim.x * blockDim.x) {
        const float v = a[i];
        c[i] = v;
        b[i] = (__bf16)v;
    }
}
__global__ void k_add_f2bf(const float* __restrict__ h, float* __restrict__ hs,
                           __bf16* __restrict__ hsb, int n) {
    for (int i = blockIdx.x * blockDim.x + threadIdx.x; i < n; i += gridDim.x * blockDim.x) {
        const float v = hs[i] + h[i];
        hs[i]  = v;
        hsb[i] = (__bf16)v;
    }
}

// ---------------------------------------------------------------------------
// Host-side orchestration
// ---------------------------------------------------------------------------
static inline void gemm(const __bf16* A, const __bf16* B, float* C,
                        int M, int N, int K, int transB, int relu, hipStream_t s)
{
    dim3 grid((N + TN - 1) / TN, (M + TM - 1) / TM);
    k_gemm_wmma<<<grid, 256, 0, s>>>(A, B, C, M, N, K, transB, relu);
}
static inline void f2bf(const float* a, __bf16* b, int n, hipStream_t s)
{
    k_f2bf<<<(n + 255) / 256, 256, 0, s>>>(a, b, n);
}

extern "C" void kernel_launch(void* const* d_in, const int* in_sizes, int n_in,
                              void* d_out, int out_size, void* d_ws, size_t ws_size,
                              hipStream_t stream)
{
    (void)in_sizes; (void)n_in; (void)out_size; (void)ws_size;
    const int N = 4096, F = 512, H1 = 256, H2 = 64, C = 16;

    const float* x      = (const float*)d_in[0];
    const float* adj0   = (const float*)d_in[1];
    const float* W_first= (const float*)d_in[2];
    const float* W_hid  = (const float*)d_in[3];
    const float* W_mu   = (const float*)d_in[4];
    const float* W_lv   = (const float*)d_in[5];
    const float* W_cls  = (const float*)d_in[6];

    float* out_adj = (float*)d_out;                       // N*N
    float* out_mu  = out_adj + (size_t)N * N;             // N*H2
    float* out_lv  = out_mu  + (size_t)N * H2;            // N*H2
    float* out_cls = out_lv  + (size_t)N * H2;            // N*C

    // workspace carve-out (256B aligned)
    char* w = (char*)d_ws;
    auto carve = [&](size_t bytes) -> char* {
        char* p = w;
        w += (bytes + 255) & ~(size_t)255;
        return p;
    };
    float*  adj_sum   = (float*) carve((size_t)N * N * 4);
    __bf16* adj_bf    = (__bf16*)carve((size_t)N * N * 2);
    float*  scores    = (float*) carve((size_t)N * N * 4);
    float*  h         = (float*) carve((size_t)N * H1 * 4);
    __bf16* h_bf      = (__bf16*)carve((size_t)N * H1 * 2);
    float*  h_sum     = (float*) carve((size_t)N * H1 * 4);
    __bf16* h_sum_bf  = (__bf16*)carve((size_t)N * H1 * 2);
    float*  t1        = (float*) carve((size_t)N * H1 * 4);
    __bf16* t1_bf     = (__bf16*)carve((size_t)N * H1 * 2);
    float*  mt        = (float*) carve((size_t)N * H2 * 4);
    __bf16* mt_bf     = (__bf16*)carve((size_t)N * H2 * 2);
    __bf16* mu_bf     = (__bf16*)carve((size_t)N * H2 * 2);
    __bf16* x_bf      = (__bf16*)carve((size_t)N * F * 2);
    __bf16* Wf_bf     = (__bf16*)carve((size_t)F * H1 * 2);
    __bf16* Wh_bf     = (__bf16*)carve((size_t)5 * H1 * H1 * 2);
    __bf16* Wmu_bf    = (__bf16*)carve((size_t)6 * H1 * H2 * 2);
    __bf16* Wlv_bf    = (__bf16*)carve((size_t)6 * H1 * H2 * 2);
    __bf16* Wc_bf     = (__bf16*)carve((size_t)H1 * C * 2);

    // ---- one-time (per call) conversions & init ----
    f2bf(x,       x_bf,   N * F,          stream);
    f2bf(W_first, Wf_bf,  F * H1,         stream);
    f2bf(W_hid,   Wh_bf,  5 * H1 * H1,    stream);
    f2bf(W_mu,    Wmu_bf, 6 * H1 * H2,    stream);
    f2bf(W_lv,    Wlv_bf, 6 * H1 * H2,    stream);
    f2bf(W_cls,   Wc_bf,  H1 * C,         stream);

    {
        const int nn = N * N;
        k_zero_f   <<<(nn + 255) / 256, 256, 0, stream>>>(out_adj, nn);
        k_copy_f2bf<<<(nn + 255) / 256, 256, 0, stream>>>(adj0, adj_sum, adj_bf, nn);
        const int nh = N * H1;
        k_zero_f <<<(nh + 255) / 256, 256, 0, stream>>>(h_sum, nh);
        k_zero_bf<<<(nh + 255) / 256, 256, 0, stream>>>(h_sum_bf, nh);
    }

    // ---- 6 VAE layers ----
    for (int i = 0; i < 6; ++i) {
        const __bf16* inp = (i == 0) ? x_bf : h_sum_bf;
        const int Kin     = (i == 0) ? F : H1;
        const __bf16* Wa  = (i == 0) ? Wf_bf : (Wh_bf + (size_t)(i - 1) * H1 * H1);

        // h = relu(adj_sum @ (inp @ Wa))
        gemm(inp, Wa, t1, N, H1, Kin, 0, 0, stream);
        f2bf(t1, t1_bf, N * H1, stream);
        gemm(adj_bf, t1_bf, h, N, H1, N, 0, 1, stream);
        f2bf(h, h_bf, N * H1, stream);

        // mu = adj_sum @ (h @ W_mu[i])
        gemm(h_bf, Wmu_bf + (size_t)i * H1 * H2, mt, N, H2, H1, 0, 0, stream);
        f2bf(mt, mt_bf, N * H2, stream);
        gemm(adj_bf, mt_bf, out_mu, N, H2, N, 0, 0, stream);
        f2bf(out_mu, mu_bf, N * H2, stream);

        // logvar only feeds the final output -> compute last iteration only
        if (i == 5) {
            gemm(h_bf, Wlv_bf + (size_t)i * H1 * H2, mt, N, H2, H1, 0, 0, stream);
            f2bf(mt, mt_bf, N * H2, stream);
            gemm(adj_bf, mt_bf, out_lv, N, H2, N, 0, 0, stream);
        }

        // scores = mu @ mu^T ; adj_i = masked softmax; accumulate
        gemm(mu_bf, mu_bf, scores, N, N, H2, 1, 0, stream);
        k_softmax_accum<<<N, 256, 0, stream>>>(scores, adj0, out_adj,
                                               adj_sum, adj_bf, N);

        // h_sum += h (next layer input)
        k_add_f2bf<<<(N * H1 + 255) / 256, 256, 0, stream>>>(h, h_sum, h_sum_bf, N * H1);
    }

    // ---- classifier: log_softmax(relu(adj_sum @ (h_sum @ W_cls))) ----
    gemm(h_sum_bf, Wc_bf, t1, N, C, H1, 0, 0, stream);
    f2bf(t1, t1_bf, N * C, stream);
    gemm(adj_bf, t1_bf, h /*reuse as cls buffer*/, N, C, N, 0, 1, stream);
    k_logsoftmax_row<<<(N + 255) / 256, 256, 0, stream>>>(h, out_cls, N, C);
}
